// Flux2JointAttention_16716012716253
// MI455X (gfx1250) — matrix operand order repaired and checked by reading, not compiled, not run on hardware
//
#include <hip/hip_runtime.h>
#include <cstdint>

#define DIM   3072
#define HEADS 24
#define HD    128
#define INNER 3072
#define SEQ   2560
#define TXT   512
#define IMG   2048

typedef __attribute__((ext_vector_type(16))) __bf16 bf16x16;
typedef __attribute__((ext_vector_type(2)))  __bf16 bf16x2;
typedef __attribute__((ext_vector_type(8)))  float  floatx8;

union BF16Frag {
  bf16x16 v;
  unsigned int w[8];
  uint4 q[2];
};

// Hardware bf16 conversion (RNE): clang emits packed v_cvt for gfx1250.
__device__ __forceinline__ unsigned int pack2_bf16(float a, float b) {
  bf16x2 v;
  v[0] = (__bf16)a;
  v[1] = (__bf16)b;
  unsigned int u;
  __builtin_memcpy(&u, &v, 4);
  return u;
}

__device__ __forceinline__ unsigned short bf16_bits(float a) {
  __bf16 h = (__bf16)a;
  unsigned short u;
  __builtin_memcpy(&u, &h, 2);
  return u;
}

#define WMMA_BF16(A, B, C) \
  __builtin_amdgcn_wmma_f32_16x16x32_bf16(false, (A), false, (B), (short)0, (C), false, false)

// CDNA5 async copy: global -> LDS without VGPR round-trip, tracked by ASYNCcnt.
__device__ __forceinline__ void async_b128(unsigned lds_off, const void* gaddr) {
  asm volatile("global_load_async_to_lds_b128 %0, %1, off"
               :: "v"(lds_off), "v"(gaddr) : "memory");
}

// LDS tile geometry: one 32-K row = 64B data + 16B pad (bank-conflict-free frags)
#define LDS_ROW 80
#define B_ROWS  256
#define A_ROWS  64
#define BUF_SZ  ((B_ROWS + A_ROWS) * LDS_ROW)   // 25600 B per stage

// ---------------------------------------------------------------------------
// Kernel 0: bulk f32 -> bf16 conversion, 8 elements per thread.
// ---------------------------------------------------------------------------
__global__ __launch_bounds__(256)
void f32_to_bf16_kernel(const float* __restrict__ src,
                        unsigned short* __restrict__ dst, int n) {
  const int i = (blockIdx.x * 256 + threadIdx.x) * 8;
  if (i + 8 > n) return;                       // grids are exact; guard is uniform
  float4 a = *(const float4*)(src + i);
  float4 b = *(const float4*)(src + i + 4);
  uint4 o;
  o.x = pack2_bf16(a.x, a.y);
  o.y = pack2_bf16(a.z, a.w);
  o.z = pack2_bf16(b.x, b.y);
  o.w = pack2_bf16(b.z, b.w);
  *(uint4*)(dst + i) = o;
}

// ---------------------------------------------------------------------------
// Shared GEMM body: async double-buffered LDS staging of A(64x32) + B(256x32),
// 8 waves, wave tile 16(M) x 128(N). K-loop unrolled x2 so the stage-buffer
// index is a compile-time constant -> fragment reads stay ds_load_b128.
// Each wave issues exactly 5 async_b128 per K-step; async loads complete in
// order, so s_wait_asynccnt 0x5 after issuing the next batch guarantees the
// current stage landed while the next streams in the background.
// Row pitch of both operands equals Kdim (3072 for every GEMM here).
// ---------------------------------------------------------------------------
__device__ __forceinline__ void gemm_body(
    const unsigned short* __restrict__ Aglob,   // block's A rows (64 x Kdim)
    const unsigned short* __restrict__ Wglob,   // block's B rows (256 x Kdim)
    int Kdim, floatx8* acc) {
  __shared__ __align__(16) unsigned char lds[2][BUF_SZ];
  const int tid  = threadIdx.x;
  const int lane = tid & 31;
  const int wave = tid >> 5;
  const int lo = lane & 15, hi = lane >> 4;

  // per-thread staging sources: thread i owns B row i (64B/step),
  // and 16B segment (i&3) of A row (i>>2).
  const unsigned short* bsrc = Wglob + (size_t)tid * Kdim;
  const unsigned short* asrc = Aglob + (size_t)(tid >> 2) * Kdim + (tid & 3) * 8;

  unsigned b_dst[2], a_dst[2];
  #pragma unroll
  for (int c = 0; c < 2; ++c) {
    b_dst[c] = (unsigned)(size_t)&lds[c][tid * LDS_ROW];
    a_dst[c] = (unsigned)(size_t)&lds[c][B_ROWS * LDS_ROW + (tid >> 2) * LDS_ROW + (tid & 3) * 16];
  }

  #pragma unroll
  for (int t = 0; t < 8; ++t) {
    #pragma unroll
    for (int r = 0; r < 8; ++r) acc[t][r] = 0.0f;
  }

  // buf is a literal 0/1 at every call site -> constant folds.
  auto stage = [&](int buf, int kk) {
    const unsigned short* bs = bsrc + kk;
    const unsigned bd = buf ? b_dst[1] : b_dst[0];
    const unsigned ad = buf ? a_dst[1] : a_dst[0];
    async_b128(bd +  0, (const void*)(bs));
    async_b128(bd + 16, (const void*)(bs + 8));
    async_b128(bd + 32, (const void*)(bs + 16));
    async_b128(bd + 48, (const void*)(bs + 24));
    async_b128(ad,      (const void*)(asrc + kk));
  };

  auto compute = [&](int buf) {
    BF16Frag af;                               // A frag: M = lo, 16 contiguous K
    const int ab = B_ROWS * LDS_ROW + ((wave >> 1) * 16 + lo) * LDS_ROW + hi * 32;
    af.q[0] = *(const uint4*)&lds[buf][ab];
    af.q[1] = *(const uint4*)&lds[buf][ab + 16];
    #pragma unroll
    for (int t = 0; t < 8; ++t) {              // B frag: N = lo, 16 contiguous K
      const int bb = ((wave & 1) * 128 + t * 16 + lo) * LDS_ROW + hi * 32;
      BF16Frag bf;
      bf.q[0] = *(const uint4*)&lds[buf][bb];
      bf.q[1] = *(const uint4*)&lds[buf][bb + 16];
      acc[t] = WMMA_BF16(af.v, bf.v, acc[t]);
    }
  };

  stage(0, 0);                                 // prologue
  for (int kk = 0; kk < Kdim; kk += 64) {      // Kdim % 64 == 0
    // half 1: compute buf0 @ kk while buf1 @ kk+32 streams in
    stage(1, kk + 32);
    asm volatile("s_wait_asynccnt 0x5" ::: "memory");
    __syncthreads();
    compute(0);
    __syncthreads();
    // half 2: compute buf1 @ kk+32 while buf0 @ kk+64 streams in
    if (kk + 64 < Kdim) {
      stage(0, kk + 64);
      asm volatile("s_wait_asynccnt 0x5" ::: "memory");
    } else {
      asm volatile("s_wait_asynccnt 0x0" ::: "memory");
    }
    __syncthreads();
    compute(1);
    __syncthreads();
  }
}

// ---------------------------------------------------------------------------
// Kernel 1: fused QKV projection. grid = (12, 40, 3), block = 256.
// A = xa (ctx rows 0..511, img rows 512..2559), f32 accum out to qkv.
// wbf layout = [wq, wk, wv, awq, awk, awv, wo, awo], each INNER*DIM.
// ---------------------------------------------------------------------------
__global__ __launch_bounds__(256)
void qkv_gemm_kernel(const unsigned short* __restrict__ xa,
                     const unsigned short* __restrict__ wbf,
                     float* __restrict__ qkv) {
  const int lane = threadIdx.x & 31;
  const int wave = threadIdx.x >> 5;
  const int lo = lane & 15, hi = lane >> 4;
  const int m0 = blockIdx.y * 64;
  const int n0 = blockIdx.x * 256;
  const int z  = blockIdx.z;
  const size_t WSZ = (size_t)INNER * DIM;
  const bool img = (m0 >= TXT);                // 512 % 64 == 0: no straddle
  const unsigned short* W = wbf + (size_t)(img ? z : z + 3) * WSZ + (size_t)n0 * DIM;
  const unsigned short* A = xa + (size_t)m0 * DIM;

  floatx8 acc[8];
  gemm_body(A, W, DIM, acc);

  float* out = qkv + (size_t)z * SEQ * INNER;
  const int mw = m0 + (wave >> 1) * 16;
  const int nw = n0 + (wave & 1) * 128;
  #pragma unroll
  for (int r = 0; r < 8; ++r) {
    const size_t row = (size_t)(mw + hi * 8 + r) * INNER + nw;
    #pragma unroll
    for (int t = 0; t < 8; ++t)
      out[row + t * 16 + lo] = acc[t][r];
  }
}

// ---------------------------------------------------------------------------
// Kernel 2: per-(s,head) RMSNorm + RoPE, bf16 pack. One wave per (s,h).
// q gets the 1/sqrt(HD) score scale folded in. v is written d-major (vt).
// ---------------------------------------------------------------------------
__global__ __launch_bounds__(256)
void norm_rope_kernel(const float* __restrict__ qkv,
                      const float* __restrict__ gq, const float* __restrict__ gk,
                      const float* __restrict__ agq, const float* __restrict__ agk,
                      const float* __restrict__ cosb, const float* __restrict__ sinb,
                      unsigned short* __restrict__ qb, unsigned short* __restrict__ kb,
                      unsigned short* __restrict__ vt) {
  const int lane = threadIdx.x & 31;
  const int wave = threadIdx.x >> 5;
  const int idx = blockIdx.x * 8 + wave;       // 0 .. 61439
  const int s = idx / HEADS, h = idx % HEADS;
  const int d0 = lane * 4;                     // lane-local rope pairs
  const size_t base = (size_t)s * INNER + h * HD + d0;

  float4 q4 = *(const float4*)(qkv + base);
  float4 k4 = *(const float4*)(qkv + (size_t)SEQ * INNER + base);
  float4 v4 = *(const float4*)(qkv + 2ull * SEQ * INNER + base);

  float sq = q4.x*q4.x + q4.y*q4.y + q4.z*q4.z + q4.w*q4.w;
  float sk = k4.x*k4.x + k4.y*k4.y + k4.z*k4.z + k4.w*k4.w;
  #pragma unroll
  for (int off = 1; off < 32; off <<= 1) {
    sq += __shfl_xor(sq, off);
    sk += __shfl_xor(sk, off);
  }
  const float rq = rsqrtf(sq * (1.0f / 128.0f) + 1e-6f);
  const float rk = rsqrtf(sk * (1.0f / 128.0f) + 1e-6f);
  const float* gQ = (s < TXT) ? agq : gq;
  const float* gK = (s < TXT) ? agk : gk;

  float qn0 = q4.x * rq * gQ[d0+0], qn1 = q4.y * rq * gQ[d0+1];
  float qn2 = q4.z * rq * gQ[d0+2], qn3 = q4.w * rq * gQ[d0+3];
  float kn0 = k4.x * rk * gK[d0+0], kn1 = k4.y * rk * gK[d0+1];
  float kn2 = k4.z * rk * gK[d0+2], kn3 = k4.w * rk * gK[d0+3];

  const float* cp = cosb + (size_t)s * HD + d0;
  const float* sp = sinb + (size_t)s * HD + d0;
  float c0 = cp[0], c1 = cp[1], c2 = cp[2], c3 = cp[3];
  float s0 = sp[0], s1 = sp[1], s2 = sp[2], s3 = sp[3];

  const float SC = 0.08838834764831845f;       // 1/sqrt(128)
  float qa = (qn0*c0 - qn1*s0) * SC, qbv = (qn1*c1 + qn0*s1) * SC;
  float qc = (qn2*c2 - qn3*s2) * SC, qd = (qn3*c3 + qn2*s3) * SC;
  float ka = kn0*c0 - kn1*s0, kbv = kn1*c1 + kn0*s1;
  float kc = kn2*c2 - kn3*s2, kd = kn3*c3 + kn2*s3;

  uint2 qo, ko;
  qo.x = pack2_bf16(qa, qbv); qo.y = pack2_bf16(qc, qd);
  ko.x = pack2_bf16(ka, kbv); ko.y = pack2_bf16(kc, kd);
  *(uint2*)(qb + base) = qo;
  *(uint2*)(kb + base) = ko;

  const size_t vb = (size_t)(h * HD + d0) * SEQ + s;
  vt[vb + 0ull*SEQ] = bf16_bits(v4.x);
  vt[vb + 1ull*SEQ] = bf16_bits(v4.y);
  vt[vb + 2ull*SEQ] = bf16_bits(v4.z);
  vt[vb + 3ull*SEQ] = bf16_bits(v4.w);
}

// ---------------------------------------------------------------------------
// Kernel 3: flash attention. grid = (40, 24), block = 128 (4 waves).
// Each wave: one head, one 16-query tile, streams keys in chunks of 32.
// Output written directly in bf16 for the projection GEMM.
// ---------------------------------------------------------------------------
__global__ __launch_bounds__(128)
void attn_kernel(const unsigned short* __restrict__ qb,
                 const unsigned short* __restrict__ kb,
                 const unsigned short* __restrict__ vt,
                 unsigned short* __restrict__ attn) {
  __shared__ float Pbuf[4][16 * 33];           // per-wave padded P staging
  const int lane = threadIdx.x & 31;
  const int wave = threadIdx.x >> 5;
  const int lo = lane & 15, hi = lane >> 4;
  const int h  = blockIdx.y;
  const int q0 = (blockIdx.x * 4 + wave) * 16;
  float* pb = Pbuf[wave];

  BF16Frag qf[4];                              // whole Q tile in registers
  #pragma unroll
  for (int kc = 0; kc < 4; ++kc) {
    const unsigned short* qp = qb + (size_t)(q0 + lo) * INNER + h * HD + kc * 32 + hi * 16;
    qf[kc].q[0] = *(const uint4*)(qp);
    qf[kc].q[1] = *(const uint4*)(qp + 8);
  }

  float m_i[8], l_i[8];
  floatx8 acc[8];
  #pragma unroll
  for (int r = 0; r < 8; ++r) { m_i[r] = -1e30f; l_i[r] = 0.0f; }
  #pragma unroll
  for (int t = 0; t < 8; ++t) {
    #pragma unroll
    for (int r = 0; r < 8; ++r) acc[t][r] = 0.0f;
  }

  for (int k0 = 0; k0 < SEQ; k0 += 32) {
    // ---- scores: two 16x16 tiles (keys k0..k0+15, k0+16..k0+31) ----
    floatx8 sA = {0,0,0,0,0,0,0,0};
    floatx8 sB = sA;
    #pragma unroll
    for (int kc = 0; kc < 4; ++kc) {
      const unsigned short* kp0 = kb + (size_t)(k0 + lo) * INNER + h * HD + kc * 32 + hi * 16;
      const unsigned short* kp1 = kp0 + (size_t)16 * INNER;
      BF16Frag f0, f1;
      f0.q[0] = *(const uint4*)(kp0); f0.q[1] = *(const uint4*)(kp0 + 8);
      f1.q[0] = *(const uint4*)(kp1); f1.q[1] = *(const uint4*)(kp1 + 8);
      sA = WMMA_BF16(qf[kc].v, f0.v, sA);
      sB = WMMA_BF16(qf[kc].v, f1.v, sB);
    }

    // ---- online softmax: row r of this lane is query (hi*8 + r) ----
    float cm[8];
    #pragma unroll
    for (int r = 0; r < 8; ++r) cm[r] = fmaxf(sA[r], sB[r]);
    #pragma unroll
    for (int off = 1; off < 16; off <<= 1) {
      #pragma unroll
      for (int r = 0; r < 8; ++r) cm[r] = fmaxf(cm[r], __shfl_xor(cm[r], off));
    }
    float alpha[8], p0[8], p1[8], rs[8];
    #pragma unroll
    for (int r = 0; r < 8; ++r) {
      const float mn = fmaxf(m_i[r], cm[r]);
      alpha[r] = __expf(m_i[r] - mn);
      p0[r] = __expf(sA[r] - mn);
      p1[r] = __expf(sB[r] - mn);
      rs[r] = p0[r] + p1[r];
      m_i[r] = mn;
    }
    #pragma unroll
    for (int off = 1; off < 16; off <<= 1) {
      #pragma unroll
      for (int r = 0; r < 8; ++r) rs[r] += __shfl_xor(rs[r], off);
    }
    #pragma unroll
    for (int r = 0; r < 8; ++r) l_i[r] = l_i[r] * alpha[r] + rs[r];
    #pragma unroll
    for (int t = 0; t < 8; ++t) {
      #pragma unroll
      for (int r = 0; r < 8; ++r) acc[t][r] *= alpha[r];
    }

    // ---- transpose P through LDS: D layout -> A layout ----
    #pragma unroll
    for (int r = 0; r < 8; ++r) {
      const int row = hi * 8 + r;
      pb[row * 33 + lo]      = p0[r];
      pb[row * 33 + 16 + lo] = p1[r];
    }
    asm volatile("s_wait_dscnt 0" ::: "memory");
    BF16Frag pa;
    const float* prow = pb + lo * 33 + hi * 8;
    #pragma unroll
    for (int j = 0; j < 4; ++j) {
      pa.w[j]     = pack2_bf16(prow[2*j],      prow[2*j + 1]);
      pa.w[4 + j] = pack2_bf16(prow[16 + 2*j], prow[16 + 2*j + 1]);
    }
    asm volatile("" ::: "memory");  // keep next-iter stores after these loads

    // ---- out += P(16x32) x V(32x128): vt is d-major so K is contiguous ----
    #pragma unroll
    for (int t = 0; t < 8; ++t) {
      const unsigned short* vp = vt + (size_t)(h * HD + t * 16 + lo) * SEQ + k0 + hi * 8;
      BF16Frag vf;
      vf.q[0] = *(const uint4*)(vp);
      vf.q[1] = *(const uint4*)(vp + 16);
      acc[t] = WMMA_BF16(pa.v, vf.v, acc[t]);
    }
  }

  #pragma unroll
  for (int r = 0; r < 8; ++r) {
    const float inv = 1.0f / l_i[r];
    const size_t row = (size_t)(q0 + hi * 8 + r) * INNER + h * HD;
    #pragma unroll
    for (int t = 0; t < 8; ++t)
      attn[row + t * 16 + lo] = bf16_bits(acc[t][r] * inv);
  }
}

// ---------------------------------------------------------------------------
// Kernel 4: output projection. grid = (12, 40), block = 256.
// img rows (s>=512) -> wo -> d_out[0..], txt rows -> awo -> d_out + IMG*DIM.
// ---------------------------------------------------------------------------
__global__ __launch_bounds__(256)
void proj_gemm_kernel(const unsigned short* __restrict__ attn,
                      const unsigned short* __restrict__ wbf,
                      float* __restrict__ out) {
  const int lane = threadIdx.x & 31;
  const int wave = threadIdx.x >> 5;
  const int lo = lane & 15, hi = lane >> 4;
  const int m0 = blockIdx.y * 64;
  const int n0 = blockIdx.x * 256;
  const size_t WSZ = (size_t)INNER * DIM;
  const bool img = (m0 >= TXT);
  const unsigned short* W = wbf + (size_t)(img ? 6 : 7) * WSZ + (size_t)n0 * DIM;
  const unsigned short* A = attn + (size_t)m0 * INNER;

  floatx8 acc[8];
  gemm_body(A, W, INNER, acc);

  const int mw = m0 + (wave >> 1) * 16;
  const int nw = n0 + (wave & 1) * 128;
  #pragma unroll
  for (int r = 0; r < 8; ++r) {
    const int m = mw + hi * 8 + r;
    float* op = img ? (out + (size_t)(m - TXT) * DIM)
                    : (out + (size_t)IMG * DIM + (size_t)m * DIM);
    #pragma unroll
    for (int t = 0; t < 8; ++t)
      op[nw + t * 16 + lo] = acc[t][r];
  }
}

// ---------------------------------------------------------------------------
extern "C" void kernel_launch(void* const* d_in, const int* in_sizes, int n_in,
                              void* d_out, int out_size, void* d_ws, size_t ws_size,
                              hipStream_t stream) {
  (void)in_sizes; (void)n_in; (void)out_size; (void)ws_size;
  const float* x        = (const float*)d_in[0];
  const float* ctx      = (const float*)d_in[1];
  const float* rope_cos = (const float*)d_in[2];
  const float* rope_sin = (const float*)d_in[3];
  const float* gq  = (const float*)d_in[7];
  const float* gk  = (const float*)d_in[8];
  const float* agq = (const float*)d_in[13];
  const float* agk = (const float*)d_in[14];
  // weight order for wbf: [wq, wk, wv, awq, awk, awv, wo, awo]
  const float* wsrc[8] = {
    (const float*)d_in[4],  (const float*)d_in[5],  (const float*)d_in[6],
    (const float*)d_in[10], (const float*)d_in[11], (const float*)d_in[12],
    (const float*)d_in[9],  (const float*)d_in[15]
  };

  const size_t WSZ = (size_t)INNER * DIM;       // 9,437,184 elements
  char* ws = (char*)d_ws;
  unsigned short* wbf = (unsigned short*)ws;                   // 8*WSZ bf16
  size_t off = 2ull * 8 * WSZ;
  unsigned short* xa = (unsigned short*)(ws + off); off += 2ull * SEQ * DIM;
  float* qkv = (float*)(ws + off);              off += 4ull * 3 * SEQ * INNER;
  unsigned short* qbuf = (unsigned short*)(ws + off); off += 2ull * SEQ * INNER;
  unsigned short* kbuf = (unsigned short*)(ws + off); off += 2ull * SEQ * INNER;
  unsigned short* vt   = (unsigned short*)(ws + off); off += 2ull * SEQ * INNER;
  unsigned short* attn = (unsigned short*)qkv;  // reuse: f32 qkv dead after norm

  // 0) one-shot bf16 conversion of weights + activations
  for (int i = 0; i < 8; ++i)
    f32_to_bf16_kernel<<<dim3((int)(WSZ / 2048)), 256, 0, stream>>>(
        wsrc[i], wbf + (size_t)i * WSZ, (int)WSZ);
  f32_to_bf16_kernel<<<dim3((TXT * DIM) / 2048), 256, 0, stream>>>(
      ctx, xa, TXT * DIM);
  f32_to_bf16_kernel<<<dim3((IMG * DIM) / 2048), 256, 0, stream>>>(
      x, xa + (size_t)TXT * DIM, IMG * DIM);

  // 1..4) pipeline
  qkv_gemm_kernel<<<dim3(12, 40, 3), 256, 0, stream>>>(xa, wbf, qkv);
  norm_rope_kernel<<<dim3(7680), 256, 0, stream>>>(
      qkv, gq, gk, agq, agk, rope_cos, rope_sin, qbuf, kbuf, vt);
  attn_kernel<<<dim3(40, 24), 128, 0, stream>>>(qbuf, kbuf, vt, attn);
  proj_gemm_kernel<<<dim3(12, 40), 256, 0, stream>>>(attn, wbf, (float*)d_out);
}
